// CrossAttentionSpatial_66202625901198
// MI455X (gfx1250) — matrix operand order repaired and checked by reading, not compiled, not run on hardware
//
#include <hip/hip_runtime.h>
#include <stdint.h>

typedef __attribute__((ext_vector_type(16))) __bf16 v16bf;
typedef __attribute__((ext_vector_type(8)))  float  v8f;
typedef __attribute__((ext_vector_type(4)))  unsigned int u32x4;
typedef __attribute__((ext_vector_type(8)))  unsigned int u32x8;

union Frag  { v16bf v; uint4 q[2]; unsigned short s[16]; };
union Pack8 { uint4 q; unsigned short s[8]; };
union Pack4 { uint2 q; unsigned short s[4]; };

__device__ __forceinline__ unsigned short f2bfu(float f) {
  __bf16 h = (__bf16)f;                      // native cvt (RNE), pairs into v_cvt_pk_bf16_f32
  return __builtin_bit_cast(unsigned short, h);
}

// ---------------- Tensor Data Mover: 2D tile (bf16 elements) -> LDS ----------------
// Descriptor per ISA cdna5 §8: group0 = {count, lds_addr, global_addr, type=2},
// group1 = {data_size=1(2B), tensor_dim0/1, tile_dim0/1, tensor_dim0_stride}.
__device__ __forceinline__ void tdm_load_2d(unsigned lds_off, const void* gptr,
                                            unsigned tensor0, unsigned tile0,
                                            unsigned tile1, unsigned long long stride0) {
  unsigned long long ga = (unsigned long long)(uintptr_t)gptr;
  u32x4 g0;
  g0[0] = 1u;                                                  // count=1, user descriptor
  g0[1] = lds_off;                                             // LDS byte offset
  g0[2] = (unsigned)ga;                                        // global_addr[31:0]
  g0[3] = ((unsigned)(ga >> 32) & 0x1FFFFFFu) | (2u << 30);    // addr[56:32] | type=2
  unsigned long long q0 = (1ull << 16)                                    // data_size=1 (2B)
                        | ((unsigned long long)(tensor0 & 0xFFFFu) << 48);// tensor_dim0[15:0]
  unsigned long long q1 = ((unsigned long long)(tensor0 >> 16) & 0xFFFFu) // tensor_dim0[31:16]
                        | ((unsigned long long)tile1 << 16)               // tensor_dim1
                        | ((unsigned long long)(tile0 & 0xFFFFu) << 48);  // tile_dim0
  unsigned long long q2 = (unsigned long long)tile1                       // tile_dim1
                        | ((stride0 & 0xFFFFFFFFull) << 32);              // dim0_stride[31:0]
  unsigned long long q3 = ((stride0 >> 32) & 0xFFFFull);                  // dim0_stride[47:32]
  u32x8 g1;
  g1[0] = (unsigned)q0; g1[1] = (unsigned)(q0 >> 32);
  g1[2] = (unsigned)q1; g1[3] = (unsigned)(q1 >> 32);
  g1[4] = (unsigned)q2; g1[5] = (unsigned)(q2 >> 32);
  g1[6] = (unsigned)q3; g1[7] = (unsigned)(q3 >> 32);
  u32x4 gz = {0u, 0u, 0u, 0u};
  asm volatile("tensor_load_to_lds %0, %1, %2, %3"
               :: "s"(g0), "s"(g1), "s"(gz), "s"(gz) : "memory");
}

__device__ __forceinline__ void wait_tensor0() {
#if __has_builtin(__builtin_amdgcn_s_wait_tensorcnt)
  __builtin_amdgcn_s_wait_tensorcnt(0);
#else
  asm volatile("s_wait_tensorcnt 0x0" ::: "memory");
#endif
}

// -------------------- fp32 -> bf16 weight pre-conversion --------------------
__global__ void cvt_bf16_kernel(const float* __restrict__ in, unsigned short* __restrict__ out,
                                int n4) {
  int i = (blockIdx.x * 256 + threadIdx.x);
  if (i < n4) {
    float4 v = *(const float4*)(in + (size_t)i * 4);
    Pack4 p;
    p.s[0] = f2bfu(v.x); p.s[1] = f2bfu(v.y); p.s[2] = f2bfu(v.z); p.s[3] = f2bfu(v.w);
    *(uint2*)(out + (size_t)i * 4) = p.q;
  }
}

// -------------------- GroupNorm (fp32 [B,C,HW] -> bf16 [B,HW,C]) --------------------
template <int CPG>
__global__ void gn_kernel(const float* __restrict__ in, const float* __restrict__ w,
                          const float* __restrict__ bias, unsigned short* __restrict__ out,
                          int C, int HW) {
  int b = blockIdx.x >> 5;          // GROUPS = 32
  int g = blockIdx.x & 31;
  size_t ibase = ((size_t)b * C + (size_t)g * CPG) * HW;
  int n = CPG * HW;
  float s = 0.f, ss = 0.f;
  for (int i = threadIdx.x; i < n; i += 256) {
    float v = in[ibase + i];
    s += v; ss += v * v;
  }
  __shared__ float red[512];
  red[threadIdx.x] = s; red[256 + threadIdx.x] = ss;
  __syncthreads();
  for (int off = 128; off > 0; off >>= 1) {
    if ((int)threadIdx.x < off) {
      red[threadIdx.x]       += red[threadIdx.x + off];
      red[256 + threadIdx.x] += red[256 + threadIdx.x + off];
    }
    __syncthreads();
  }
  float mu   = red[0] / (float)n;
  float var  = red[256] / (float)n - mu * mu;
  float rstd = rsqrtf(var + 1e-5f);
  float wv[CPG], bv[CPG];
  #pragma unroll
  for (int c = 0; c < CPG; ++c) {
    float wc = w[g * CPG + c];
    wv[c] = wc * rstd;
    bv[c] = bias[g * CPG + c] - mu * rstd * wc;
  }
  unsigned short* ob = out + (size_t)b * HW * C + g * CPG;
  for (int pix = threadIdx.x; pix < HW; pix += 256) {
    Frag pk;
    #pragma unroll
    for (int c = 0; c < CPG; ++c)
      pk.s[c] = f2bfu(in[ibase + (size_t)c * HW + pix] * wv[c] + bv[c]);
    uint4* dst = (uint4*)(ob + (size_t)pix * C);
    dst[0] = pk.q[0];
    if (CPG == 16) dst[1] = pk.q[1];
  }
}

// -------------------- 1x1 conv: Y[o,n] = Wb[o,c] @ Xt[n,c] + bias --------------------
// Wb is pre-converted bf16 [Cout,Cin]. transposed=1 -> Y stored [N,Cout], else [Cout,N].
__global__ void conv1x1_wmma(const unsigned short* __restrict__ Wb, const float* __restrict__ bias,
                             const unsigned short* __restrict__ Xt, unsigned short* __restrict__ Y,
                             int Cout, int Cin, int N, int transposed) {
  int wave = threadIdx.x >> 5;
  int lane = threadIdx.x & 31;
  int b = blockIdx.y;
  int tiles_n = N >> 4;
  int tile = blockIdx.x * 8 + wave;
  int ot = tile / tiles_n, nt = tile % tiles_n;
  int o0 = ot << 4, n0 = nt << 4;
  int m = lane & 15;
  int half8 = (lane & 16) ? 8 : 0;
  int hsel  = (lane & 16) ? 16 : 0;

  v8f acc = {};
  const unsigned short* wrow = Wb + (size_t)(o0 + m) * Cin;
  const unsigned short* xrow = Xt + ((size_t)b * N + n0 + m) * Cin;
  for (int c0 = 0; c0 < Cin; c0 += 32) {
    Frag a, bb;
    a.q[0] = *(const uint4*)(wrow + c0 + half8);
    a.q[1] = *(const uint4*)(wrow + c0 + half8 + 16);
    const uint4* xp = (const uint4*)(xrow + c0 + half8);
    bb.q[0] = xp[0]; bb.q[1] = xp[2];
    acc = __builtin_amdgcn_wmma_f32_16x16x32_bf16(false, a.v, false, bb.v,
                                                  (short)0, acc, false, false);
  }
  unsigned short* Yb = Y + (size_t)b * (size_t)Cout * N;
  if (transposed) {
    float4 b0 = *(const float4*)(bias + o0 + half8);
    float4 b1 = *(const float4*)(bias + o0 + half8 + 4);
    Pack8 pk;
    pk.s[0]=f2bfu(acc[0]+b0.x); pk.s[1]=f2bfu(acc[1]+b0.y);
    pk.s[2]=f2bfu(acc[2]+b0.z); pk.s[3]=f2bfu(acc[3]+b0.w);
    pk.s[4]=f2bfu(acc[4]+b1.x); pk.s[5]=f2bfu(acc[5]+b1.y);
    pk.s[6]=f2bfu(acc[6]+b1.z); pk.s[7]=f2bfu(acc[7]+b1.w);
    *(uint4*)(Yb + (size_t)(n0 + m) * Cout + o0 + half8) = pk.q;
  } else {
    #pragma unroll
    for (int r = 0; r < 8; ++r) {
      int mg = o0 + r + half8;
      Yb[(size_t)mg * N + n0 + m] = f2bfu(acc[r] + bias[mg]);
    }
  }
}

// -------------------- Flash cross-attention --------------------
// Q:[B,N,C] bf16, Kt:[B,M,C] bf16, V:[B,C,M] bf16 -> Out:[B,C,N] fp32
__global__ void __launch_bounds__(256, 1)
attn_kernel(const unsigned short* __restrict__ Q, const unsigned short* __restrict__ Kt,
            const unsigned short* __restrict__ V, float* __restrict__ Out,
            int C, int N, int M) {
  __shared__ unsigned short lds_k[32 * 256];     // [m][c]  (TDM 1-D copy)
  __shared__ unsigned short lds_v[256 * 32];     // [c][m]  (TDM 2-D tile)
  __shared__ unsigned short lds_p[8 * 16 * 32];  // per-wave P scratch, bf16

  int wave = threadIdx.x >> 5, lane = threadIdx.x & 31;
  int b = blockIdx.y;
  int qbase = blockIdx.x * 128 + wave * 16;
  int mcol = lane & 15;
  int half8 = (lane & 16) ? 8 : 0;
  int hsel  = (lane & 16) ? 16 : 0;

  // Q A-fragments: 16 queries x 256 channels, two b128 loads per fragment
  v16bf qa[8];
  const unsigned short* Qrow = Q + ((size_t)b * N + qbase + mcol) * C;
  #pragma unroll
  for (int cs = 0; cs < 8; ++cs) {
    Frag f;
    f.q[0] = *(const uint4*)(Qrow + cs * 32 + half8);
    f.q[1] = *(const uint4*)(Qrow + cs * 32 + half8 + 16);
    qa[cs] = f.v;
  }

  v8f zero = {};
  v8f o[16];
  #pragma unroll
  for (int t = 0; t < 16; ++t) o[t] = zero;
  float my_m[8], my_l[8];
  #pragma unroll
  for (int r = 0; r < 8; ++r) { my_m[r] = -1e30f; my_l[r] = 0.f; }

  const unsigned short* Kb = Kt + (size_t)b * (size_t)M * C;
  const unsigned short* Vb = V + (size_t)b * (size_t)C * M;
  unsigned ldsk_off = (unsigned)(uintptr_t)(void*)lds_k;
  unsigned ldsv_off = (unsigned)(uintptr_t)(void*)lds_v;
  const float inv_scale = 0.0625f;  // 1/sqrt(C), C=256

  for (int m0 = 0; m0 < M; m0 += 32) {
    __syncthreads();                 // previous tile fully consumed
    if (wave == 0) {
      // K rows m0..m0+31 of [M,C]: fully contiguous -> 1-D, 8192 bf16
      tdm_load_2d(ldsk_off, Kb + (size_t)m0 * C, 8192u, 8192u, 1u, 8192ull);
      // V tile: 256 rows (c) x 32 cols (m), row stride M
      tdm_load_2d(ldsv_off, Vb + m0, 32u, 32u, 256u, (unsigned long long)M);
      wait_tensor0();
    }
    __syncthreads();

    // S = Q K^T : two 16x16 n-tiles, K-loop over 256 channels
    v8f s0 = zero, s1 = zero;
    #pragma unroll
    for (int cs = 0; cs < 8; ++cs) {
      Frag f;
      const uint4* p0 = (const uint4*)(lds_k + (size_t)mcol * 256 + cs * 32 + half8);
      f.q[0] = p0[0]; f.q[1] = p0[2];
      s0 = __builtin_amdgcn_wmma_f32_16x16x32_bf16(false, qa[cs], false, f.v,
                                                   (short)0, s0, false, false);
      const uint4* p1 = (const uint4*)(lds_k + (size_t)(16 + mcol) * 256 + cs * 32 + half8);
      f.q[0] = p1[0]; f.q[1] = p1[2];
      s1 = __builtin_amdgcn_wmma_f32_16x16x32_bf16(false, qa[cs], false, f.v,
                                                   (short)0, s1, false, false);
    }

    // online softmax (row stats are half-local in the C/D layout)
    float alpha[8];
    #pragma unroll
    for (int r = 0; r < 8; ++r) {
      float x0 = s0[r] * inv_scale, x1 = s1[r] * inv_scale;
      float mx = fmaxf(x0, x1);
      #pragma unroll
      for (int d = 1; d < 16; d <<= 1) mx = fmaxf(mx, __shfl_xor(mx, d, 32));
      float nm = fmaxf(my_m[r], mx);
      float p0 = __expf(x0 - nm), p1 = __expf(x1 - nm);
      float rs = p0 + p1;
      #pragma unroll
      for (int d = 1; d < 16; d <<= 1) rs += __shfl_xor(rs, d, 32);
      float al = __expf(my_m[r] - nm);
      my_l[r] = my_l[r] * al + rs;
      my_m[r] = nm;
      alpha[r] = al;
      s0[r] = p0; s1[r] = p1;
    }
    #pragma unroll
    for (int t = 0; t < 16; ++t) {
      #pragma unroll
      for (int r = 0; r < 8; ++r) o[t][r] *= alpha[r];
    }

    // P: C/D layout -> A-fragment layout via per-wave LDS scratch (bf16)
    unsigned short* pw = lds_p + wave * 512;
    #pragma unroll
    for (int r = 0; r < 8; ++r) {
      pw[(r + half8) * 32 + mcol]      = f2bfu(s0[r]);
      pw[(r + half8) * 32 + 16 + mcol] = f2bfu(s1[r]);
    }
    asm volatile("s_wait_dscnt 0" ::: "memory");
    Frag pa;
    pa.q[0] = *(const uint4*)(pw + mcol * 32 + half8);
    pa.q[1] = *(const uint4*)(pw + mcol * 32 + half8 + 16);

    // O += P V : 16 c-tiles
    #pragma unroll
    for (int t = 0; t < 16; ++t) {
      Frag f;
      const uint4* p = (const uint4*)(lds_v + (size_t)(t * 16 + mcol) * 32 + half8);
      f.q[0] = p[0]; f.q[1] = p[2];
      o[t] = __builtin_amdgcn_wmma_f32_16x16x32_bf16(false, pa.v, false, f.v,
                                                     (short)0, o[t], false, false);
    }
  }

  // epilogue: rows are contiguous per lane -> 2x float4 stores per tile
  float inv_l[8];
  #pragma unroll
  for (int r = 0; r < 8; ++r) inv_l[r] = 1.0f / my_l[r];
  #pragma unroll
  for (int t = 0; t < 16; ++t) {
    float4 lo = { o[t][0]*inv_l[0], o[t][1]*inv_l[1], o[t][2]*inv_l[2], o[t][3]*inv_l[3] };
    float4 hi = { o[t][4]*inv_l[4], o[t][5]*inv_l[5], o[t][6]*inv_l[6], o[t][7]*inv_l[7] };
    float* dst = Out + ((size_t)b * C + t * 16 + mcol) * N + qbase + half8;
    *(float4*)dst = lo;
    *(float4*)(dst + 4) = hi;
  }
}

// -------------------- host launch --------------------
extern "C" void kernel_launch(void* const* d_in, const int* in_sizes, int n_in,
                              void* d_out, int out_size, void* d_ws, size_t ws_size,
                              hipStream_t stream) {
  (void)in_sizes; (void)n_in; (void)out_size; (void)ws_size;
  const float* x      = (const float*)d_in[0];
  const float* condA  = (const float*)d_in[1];
  const float* gn_x_w = (const float*)d_in[2];
  const float* gn_x_b = (const float*)d_in[3];
  const float* gn_c_w = (const float*)d_in[4];
  const float* gn_c_b = (const float*)d_in[5];
  const float* q_w    = (const float*)d_in[6];
  const float* q_b    = (const float*)d_in[7];
  const float* k_w    = (const float*)d_in[8];
  const float* k_b    = (const float*)d_in[9];
  const float* v_w    = (const float*)d_in[10];
  const float* v_b    = (const float*)d_in[11];
  float* out = (float*)d_out;

  const int B = 8, C = 256, E = 512, HW = 4096;
  unsigned short* ws = (unsigned short*)d_ws;
  unsigned short* h   = ws;                          // [B,HW,C]  bf16
  unsigned short* h1  = h   + (size_t)B * C * HW;    // [B,HW,E]  bf16
  unsigned short* qb  = h1  + (size_t)B * E * HW;    // [B,HW,C]  bf16
  unsigned short* kb  = qb  + (size_t)B * C * HW;    // [B,HW,C]  bf16 (K^T)
  unsigned short* vb  = kb  + (size_t)B * C * HW;    // [B,C,HW]  bf16
  unsigned short* wqb = vb  + (size_t)B * C * HW;    // [C,C]     bf16
  unsigned short* wkb = wqb + (size_t)C * C;         // [C,E]     bf16
  unsigned short* wvb = wkb + (size_t)C * E;         // [C,E]     bf16

  cvt_bf16_kernel<<<dim3((C * C / 4 + 255) / 256), 256, 0, stream>>>(q_w, wqb, C * C / 4);
  cvt_bf16_kernel<<<dim3((C * E / 4 + 255) / 256), 256, 0, stream>>>(k_w, wkb, C * E / 4);
  cvt_bf16_kernel<<<dim3((C * E / 4 + 255) / 256), 256, 0, stream>>>(v_w, wvb, C * E / 4);

  gn_kernel<8><<<dim3(B * 32), 256, 0, stream>>>(x,     gn_x_w, gn_x_b, h,  C, HW);
  gn_kernel<16><<<dim3(B * 32), 256, 0, stream>>>(condA, gn_c_w, gn_c_b, h1, E, HW);

  int conv_blocks = (C / 16) * (HW / 16) / 8;   // 512
  conv1x1_wmma<<<dim3(conv_blocks, B), 256, 0, stream>>>(wqb, q_b, h,  qb, C, C, HW, 1);
  conv1x1_wmma<<<dim3(conv_blocks, B), 256, 0, stream>>>(wkb, k_b, h1, kb, C, E, HW, 1);
  conv1x1_wmma<<<dim3(conv_blocks, B), 256, 0, stream>>>(wvb, v_b, h1, vb, C, E, HW, 0);

  attn_kernel<<<dim3(HW / 128, B), 256, 0, stream>>>(qb, kb, vb, out, C, HW, HW);
}
